// baceModel_27006754358099
// MI455X (gfx1250) — compile-verified, matrix operand order checked by reading, not run on hardware
//
#include <hip/hip_runtime.h>
#include <hip/hip_bf16.h>
#include <math.h>
#include <stdint.h>

// ---------------- problem constants (from reference) ----------------
#define NN     50000        // nodes
#define EE     400000       // edges
#define GG     100          // graphs
#define NPER   500          // nodes per graph
#define KP1    200          // kept after pool1
#define KP2    80           // kept after pool2
#define N1     (GG*KP1)     // 20000
#define CD     64           // channel dim

typedef __attribute__((ext_vector_type(16))) _Float16 v16h;
typedef __attribute__((ext_vector_type(8)))  float    v8f;

// order-preserving float<->u32 map for atomic float max
__device__ __forceinline__ unsigned f32_ord(float f) {
  int i = __float_as_int(f);
  return (i >= 0) ? ((unsigned)i | 0x80000000u) : ~((unsigned)i);
}
__device__ __forceinline__ float ord_f32(unsigned u) {
  return (u & 0x80000000u) ? __int_as_float((int)(u & 0x7fffffffu))
                           : __int_as_float((int)~u);
}

// ---------------- fills ----------------
__global__ void fill_f32_k(float* p, float v, size_t n) {
  size_t i = (size_t)blockIdx.x * 256 + threadIdx.x;
  if (i < n) p[i] = v;
}
__global__ void fill_u32_k(unsigned* p, unsigned v, size_t n) {
  size_t i = (size_t)blockIdx.x * 256 + threadIdx.x;
  if (i < n) p[i] = v;
}
__global__ void fill_i32_k(int* p, int v, size_t n) {
  size_t i = (size_t)blockIdx.x * 256 + threadIdx.x;
  if (i < n) p[i] = v;
}

// ---------------- WMMA GEMM: D[M,NC] = act(A[M,K] (+abias)) x B[K,NC] (+bias) ----------------
// One wave -> one 16-row x NC strip, K-loop of v_wmma_f32_16x16x32_f16.
// B is staged in LDS *pre-swizzled into fragment order*: the 16 f16 values a lane
// needs for one (kt, half, ncol) B-fragment are contiguous (32B) -> 2x ds_load_b128.
// Fragment slot (kt*2+hb)*NC + n, element i maps to K = kt*32 + hb*8 + (i<8 ? i : i+8)
// per ISA 7.12.2:
//   A frag (16x32 f16): lane l holds row m=l&15; elems 0..7 = K kb..kb+7, 8..15 = K kb+16..kb+23, kb = l<16?0:8
//   B frag (32x16 f16): lane l holds col n=l&15; same K indexing as A rows
//   C/D (16x16 f32): lane l col n=l&15; elem r -> row r + (l<16?0:8)
template <int K, int NC, bool TANH_A, bool HAS_ABIAS>
__global__ __launch_bounds__(256) void gemm_wmma(
    const float* __restrict__ A, const float* __restrict__ abias,
    const float* __restrict__ B, const float* __restrict__ bias,
    float* __restrict__ D, int M) {
  static_assert(K % 32 == 0 && NC % 16 == 0, "shape");
  constexpr int NFRAG = (K / 32) * 2 * NC;       // v16h fragments in LDS
  __shared__ v16h sBv[NFRAG];
  _Float16* sB = (_Float16*)sBv;

  // cooperative staging: swizzle row-major B[K][NC] into fragment order (f16)
  for (int idx = threadIdx.x; idx < K * NC; idx += 256) {
    int i    = idx & 15;
    int slot = idx >> 4;
    int n    = slot % NC;
    int ktb  = slot / NC;          // kt*2 + hb
    int kt   = ktb >> 1;
    int hb   = ktb & 1;
    int k    = kt * 32 + hb * 8 + ((i < 8) ? i : (i + 8));
    sB[idx] = (_Float16)B[(size_t)k * NC + n];
  }
  __syncthreads();

  const int wave = threadIdx.x >> 5;
  const int lane = threadIdx.x & 31;
  const int mt   = blockIdx.x * 8 + wave;
  const int row  = mt * 16 + (lane & 15);
  const bool rowOK = (row < M);
  const int kb_lo = (lane < 16) ? 0 : 8;
  const int hb    = (lane < 16) ? 0 : 1;

  constexpr int NT = NC / 16;
  v8f acc[NT];
  #pragma unroll
  for (int nt = 0; nt < NT; ++nt) { v8f z = {}; acc[nt] = z; }

  const float* arow = A + (size_t)row * K;
  #pragma unroll
  for (int kt = 0; kt < K / 32; ++kt) {
    const int k0 = kt * 32 + kb_lo;
    v16h afrag;
    if (rowOK) {
      __builtin_prefetch(arow + k0 + 32, 0, 1);   // global_prefetch_b8 (speculative)
      #pragma unroll
      for (int i = 0; i < 8; ++i) {
        float a0 = arow[k0 + i];
        float a1 = arow[k0 + 16 + i];
        if (HAS_ABIAS) { a0 += abias[k0 + i]; a1 += abias[k0 + 16 + i]; }
        if (TANH_A)    { a0 = tanhf(a0);      a1 = tanhf(a1); }
        afrag[i]     = (_Float16)a0;
        afrag[8 + i] = (_Float16)a1;
      }
    } else {
      #pragma unroll
      for (int i = 0; i < 16; ++i) afrag[i] = (_Float16)0.0f;
    }
    const int slotbase = (kt * 2 + hb) * NC + (lane & 15);
    #pragma unroll
    for (int nt = 0; nt < NT; ++nt) {
      v16h bfrag = sBv[slotbase + nt * 16];       // 32B contiguous -> 2x ds_load_b128
      acc[nt] = __builtin_amdgcn_wmma_f32_16x16x32_f16(
          false, afrag, false, bfrag, (short)0, acc[nt], false, false);
    }
  }

  const int ncolbase = lane & 15;
  const int mofs = (lane < 16) ? 0 : 8;
  #pragma unroll
  for (int nt = 0; nt < NT; ++nt) {
    const int col = nt * 16 + ncolbase;
    const float bv = bias ? bias[col] : 0.0f;
    #pragma unroll
    for (int r = 0; r < 8; ++r) {
      const int m = mt * 16 + mofs + r;
      if (m < M) D[(size_t)m * NC + col] = acc[nt][r] + bv;
    }
  }
}

// ---------------- GCN ----------------
__global__ void deg_kernel(const int* __restrict__ dst, float* deg, int E) {
  int e = blockIdx.x * 256 + threadIdx.x;
  if (e < E) atomicAdd(&deg[dst[e]], 1.0f);
}
__global__ void rsqrt_kernel(float* d, int n) {
  int i = blockIdx.x * 256 + threadIdx.x;
  if (i < n) d[i] = rsqrtf(fmaxf(d[i], 1.0f));
}
__global__ void gcn_scatter(const int* __restrict__ src, const int* __restrict__ dst,
                            const float* __restrict__ dinv, const float* __restrict__ hW,
                            float* out, int E) {
  size_t gid = (size_t)blockIdx.x * 256 + threadIdx.x;
  if (gid >= (size_t)E * 64) return;
  int e = (int)(gid >> 6), c = (int)(gid & 63);
  int s = src[e], d = dst[e];
  float nrm = dinv[s] * dinv[d];
  atomicAdd(&out[(size_t)d * 64 + c], nrm * hW[(size_t)s * 64 + c]);
}
__global__ void gcn_self(const float* __restrict__ dinv, const float* __restrict__ hW,
                         const float* __restrict__ b, float* out, int n) {
  size_t gid = (size_t)blockIdx.x * 256 + threadIdx.x;
  if (gid >= (size_t)n * 64) return;
  int i = (int)(gid >> 6), c = (int)(gid & 63);
  out[(size_t)i * 64 + c] += dinv[i] * dinv[i] * hW[(size_t)i * 64 + c] + b[c];
}

// ---------------- GATv2 score: a^T leaky_relu(xl[s] + xr[d]) ----------------
__global__ void gat_score(const int* __restrict__ src, const int* __restrict__ dst,
                          const float* __restrict__ xl, const float* __restrict__ xr,
                          const float* __restrict__ att, float* score,
                          int E, int nSelf, int H) {
  size_t gid = (size_t)blockIdx.x * 256 + threadIdx.x;
  size_t tot = (size_t)(E + nSelf) * H;
  if (gid >= tot) return;
  int h = (int)(gid % H), e = (int)(gid / H);
  int s, d;
  if (e < E) { s = src[e]; d = dst[e]; } else { s = d = e - E; }
  size_t stride = (size_t)H * 64;
  const float* pl = xl + (size_t)s * stride + (size_t)h * 64;
  const float* pr = xr + (size_t)d * stride + (size_t)h * 64;
  const float* pa = att + (size_t)h * 64;
  float acc = 0.0f;
  #pragma unroll 8
  for (int c = 0; c < 64; ++c) {
    float v = pl[c] + pr[c];
    v = (v > 0.0f) ? v : 0.2f * v;
    acc = fmaf(pa[c], v, acc);
  }
  score[(size_t)e * H + h] = acc;
}

// ---------------- segment softmax passes ----------------
__global__ void seg_max_k(const int* __restrict__ dst, const float* __restrict__ score,
                          unsigned* smax, int E, int nSelf, int H) {
  size_t gid = (size_t)blockIdx.x * 256 + threadIdx.x;
  size_t tot = (size_t)(E + nSelf) * H;
  if (gid >= tot) return;
  int h = (int)(gid % H), e = (int)(gid / H);
  int d = (e < E) ? dst[e] : (e - E);
  atomicMax(&smax[(size_t)d * H + h], f32_ord(score[(size_t)e * H + h]));
}
__global__ void seg_expsum_k(const int* __restrict__ dst, float* score,
                             const unsigned* __restrict__ smax, float* ssum,
                             int E, int nSelf, int H) {
  size_t gid = (size_t)blockIdx.x * 256 + threadIdx.x;
  size_t tot = (size_t)(E + nSelf) * H;
  if (gid >= tot) return;
  int h = (int)(gid % H), e = (int)(gid / H);
  int d = (e < E) ? dst[e] : (e - E);
  size_t idx = (size_t)e * H + h;
  float ex = expf(score[idx] - ord_f32(smax[(size_t)d * H + h]));
  score[idx] = ex;
  atomicAdd(&ssum[(size_t)d * H + h], ex);
}
__global__ void seg_aggregate_k(const int* __restrict__ src, const int* __restrict__ dst,
                                const float* __restrict__ score, const float* __restrict__ ssum,
                                const float* __restrict__ msgsrc, float* out,
                                int E, int nSelf, int H) {
  size_t gid = (size_t)blockIdx.x * 256 + threadIdx.x;
  size_t tot = (size_t)(E + nSelf) * H * 64;
  if (gid >= tot) return;
  int c = (int)(gid & 63);
  size_t eh = gid >> 6;
  int h = (int)(eh % H), e = (int)(eh / H);
  int s, d;
  if (e < E) { s = src[e]; d = dst[e]; } else { s = d = e - E; }
  float alpha = score[(size_t)e * H + h] / (ssum[(size_t)d * H + h] + 1e-16f);
  size_t stride = (size_t)H * 64;
  atomicAdd(&out[(size_t)d * stride + (size_t)h * 64 + c],
            alpha * msgsrc[(size_t)s * stride + (size_t)h * 64 + c]);
}

// ---------------- transformer: score = <q[dst],k[src]>/sqrt(64) ----------------
__global__ void trans_score_k(const int* __restrict__ src, const int* __restrict__ dst,
                              const float* __restrict__ q, const float* __restrict__ k,
                              float* score, int E, int H) {
  size_t gid = (size_t)blockIdx.x * 256 + threadIdx.x;
  size_t tot = (size_t)E * H;
  if (gid >= tot) return;
  int h = (int)(gid % H), e = (int)(gid / H);
  const float* pq = q + (size_t)dst[e] * (H * 64) + (size_t)h * 64;
  const float* pk = k + (size_t)src[e] * (H * 64) + (size_t)h * 64;
  float acc = 0.0f;
  #pragma unroll 8
  for (int c = 0; c < 64; ++c) acc = fmaf(pq[c], pk[c], acc);
  score[(size_t)e * H + h] = acc * 0.125f;   // 1/sqrt(64)
}

// beta = sigmoid([msg, r, msg-r] @ wbeta); out = beta*r + (1-beta)*msg  (wave per node)
__global__ __launch_bounds__(256) void trans_beta_k(const float* __restrict__ msg,
                                                    const float* __restrict__ r,
                                                    const float* __restrict__ wbeta,
                                                    float* out, int n) {
  int node = blockIdx.x * 8 + (threadIdx.x >> 5);
  int lane = threadIdx.x & 31;
  if (node >= n) return;
  const float* pm = msg + (size_t)node * 128;
  const float* pr = r   + (size_t)node * 128;
  float acc = 0.0f;
  #pragma unroll
  for (int c = lane; c < 128; c += 32) {
    float m = pm[c], rr = pr[c];
    acc += m * wbeta[c] + rr * wbeta[128 + c] + (m - rr) * wbeta[256 + c];
  }
  #pragma unroll
  for (int off = 16; off > 0; off >>= 1) acc += __shfl_xor(acc, off, 32);
  float beta = 1.0f / (1.0f + expf(-acc));
  #pragma unroll
  for (int c = lane; c < 128; c += 32)
    out[(size_t)node * 128 + c] = beta * pr[c] + (1.0f - beta) * pm[c];
}

// ---------------- GraphNorm (block per graph, thread per column) ----------------
__global__ __launch_bounds__(64) void graph_norm_k(const float* __restrict__ x,
                                                   const float* __restrict__ w,
                                                   const float* __restrict__ b,
                                                   const float* __restrict__ ms,
                                                   float* out, int nper) {
  int g = blockIdx.x, c = threadIdx.x;
  const float* px = x + (size_t)g * nper * 64;
  float sum = 0.0f;
  for (int i = 0; i < nper; ++i) sum += px[(size_t)i * 64 + c];
  float sh = ms[c] * (sum / (float)nper);
  float var = 0.0f;
  for (int i = 0; i < nper; ++i) { float s = px[(size_t)i * 64 + c] - sh; var += s * s; }
  float inv = rsqrtf(var / (float)nper + 2e-5f);
  float* po = out + (size_t)g * nper * 64;
  for (int i = 0; i < nper; ++i)
    po[(size_t)i * 64 + c] = w[c] * (px[(size_t)i * 64 + c] - sh) * inv + b[c];
}

// ---------------- TopK pool: bitonic sort per graph in LDS ----------------
template <int SIZE>
__global__ __launch_bounds__(SIZE) void topk_pool(const float* __restrict__ x,
                                                  const float* __restrict__ w,
                                                  float* xout, int* perm,
                                                  int nper, int k) {
  __shared__ float ss[SIZE];
  __shared__ int   si[SIZE];
  __shared__ float wn;
  int g = blockIdx.x, t = threadIdx.x;
  if (t == 0) {
    float a = 0.0f;
    for (int c = 0; c < 64; ++c) a += w[c] * w[c];
    wn = 1.0f / sqrtf(a);
  }
  __syncthreads();
  float sc = -INFINITY;
  if (t < nper) {
    const float* px = x + ((size_t)g * nper + t) * 64;
    float a = 0.0f;
    #pragma unroll 8
    for (int c = 0; c < 64; ++c) a = fmaf(px[c], w[c], a);
    sc = tanhf(a * wn);
  }
  ss[t] = sc; si[t] = t;
  __syncthreads();
  // bitonic sort: descending score, ascending idx on ties (matches lax.top_k)
  for (int ksz = 2; ksz <= SIZE; ksz <<= 1) {
    for (int j = ksz >> 1; j > 0; j >>= 1) {
      int ixj = t ^ j;
      if (ixj > t) {
        float s1 = ss[t], s2 = ss[ixj];
        int   i1 = si[t], i2 = si[ixj];
        bool desc = ((t & ksz) == 0);
        bool bad  = desc ? ((s1 < s2) || (s1 == s2 && i1 > i2))
                         : ((s1 > s2) || (s1 == s2 && i1 < i2));
        if (bad) { ss[t] = s2; ss[ixj] = s1; si[t] = i2; si[ixj] = i1; }
      }
      __syncthreads();
    }
  }
  if (t < k) {
    int node = g * nper + si[t];
    if (perm) perm[g * k + t] = node;
    float v = ss[t];
    const float* px = x + (size_t)node * 64;
    float* po = xout + ((size_t)g * k + t) * 64;
    for (int c = 0; c < 64; ++c) po[c] = px[c] * v;
  }
}

// ---------------- edge filtering ----------------
__global__ void build_newid_k(const int* __restrict__ perm, int* new_id, int n1) {
  int j = blockIdx.x * 256 + threadIdx.x;
  if (j < n1) new_id[perm[j]] = j;
}
__global__ void remap_edges_k(const int* __restrict__ src, const int* __restrict__ dst,
                              const int* __restrict__ new_id, int* s1, int* d1,
                              int E, int n1) {
  int e = blockIdx.x * 256 + threadIdx.x;
  if (e >= E) return;
  int ns = new_id[src[e]], nd = new_id[dst[e]];
  bool valid = (ns < n1) && (nd < n1);
  s1[e] = valid ? ns : n1;
  d1[e] = valid ? nd : n1;
}

// ---------------- readout ----------------
__global__ __launch_bounds__(64) void readout_k(const float* __restrict__ h2,
                                                float* feats, int k2) {
  int g = blockIdx.x, c = threadIdx.x;
  const float* p = h2 + (size_t)g * k2 * 64;
  float sum = 0.0f, mx = -INFINITY;
  for (int i = 0; i < k2; ++i) {
    float v = p[(size_t)i * 64 + c];
    sum += v; mx = fmaxf(mx, v);
  }
  feats[(size_t)g * 192 + c]       = sum / (float)k2;  // mean
  feats[(size_t)g * 192 + 64 + c]  = sum;              // add
  feats[(size_t)g * 192 + 128 + c] = mx;               // max
}
__global__ void final_out_k(const float* __restrict__ feats, const float* __restrict__ ow,
                            const float* __restrict__ ob, float* out, int Gn) {
  int g = blockIdx.x * blockDim.x + threadIdx.x;
  if (g >= Gn) return;
  float a = ob[0];
  const float* f = feats + (size_t)g * 192;
  #pragma unroll 8
  for (int i = 0; i < 192; ++i) a = fmaf(f[i], ow[i], a);
  out[g] = a;
}

// ---------------- host orchestration ----------------
static inline unsigned nb256(size_t n) { return (unsigned)((n + 255) / 256); }

extern "C" void kernel_launch(void* const* d_in, const int* in_sizes, int n_in,
                              void* d_out, int out_size, void* d_ws, size_t ws_size,
                              hipStream_t stream) {
  const float* x      = (const float*)d_in[0];
  const int*   eidx   = (const int*)d_in[1];
  const int*   src    = eidx;
  const int*   dst    = eidx + EE;
  const float* gcn_w  = (const float*)d_in[3];
  const float* gcn_b  = (const float*)d_in[4];
  const float* g1_wl  = (const float*)d_in[5];
  const float* g1_wr  = (const float*)d_in[6];
  const float* g1_att = (const float*)d_in[7];
  const float* g1_b   = (const float*)d_in[8];
  const float* lin1_w = (const float*)d_in[9];
  const float* lin1_b = (const float*)d_in[10];
  const float* t_wq   = (const float*)d_in[11];
  const float* t_wk   = (const float*)d_in[12];
  const float* t_wv   = (const float*)d_in[13];
  const float* t_wsk  = (const float*)d_in[14];
  const float* t_bsk  = (const float*)d_in[15];
  const float* t_wb   = (const float*)d_in[16];
  const float* lin2_w = (const float*)d_in[17];
  const float* lin2_b = (const float*)d_in[18];
  const float* gn_w   = (const float*)d_in[19];
  const float* gn_b   = (const float*)d_in[20];
  const float* gn_ms  = (const float*)d_in[21];
  const float* p1_w   = (const float*)d_in[22];
  const float* p2_w   = (const float*)d_in[23];
  const float* g3_wl  = (const float*)d_in[24];
  const float* g3_wr  = (const float*)d_in[25];
  const float* g3_att = (const float*)d_in[26];
  const float* g3_b   = (const float*)d_in[27];
  const float* lin3_w = (const float*)d_in[28];
  const float* lin3_b = (const float*)d_in[29];
  const float* out_w  = (const float*)d_in[30];
  const float* out_b  = (const float*)d_in[31];

  float* out   = (float*)d_out;        // [100] predictions
  float* feats = out + GG;             // [100 x 192]

  // bump allocator over workspace
  uintptr_t cur = (uintptr_t)d_ws;
  auto alloc = [&](size_t bytes) -> void* {
    void* p = (void*)cur;
    cur += (bytes + 255) & ~(size_t)255;
    return p;
  };
  float*    f_hA   = (float*)alloc((size_t)NN * 64 * 4);
  float*    f_h0   = (float*)alloc((size_t)NN * 64 * 4);
  float*    f_hB   = (float*)alloc((size_t)NN * 64 * 4);
  float*    big1   = (float*)alloc((size_t)NN * 256 * 4);
  float*    big2   = (float*)alloc((size_t)NN * 256 * 4);
  float*    big3   = (float*)alloc((size_t)NN * 256 * 4);
  float*    f_sc   = (float*)alloc((size_t)(EE + NN) * 4 * 4);
  unsigned* u_smax = (unsigned*)alloc((size_t)NN * 4 * 4);
  float*    f_ssum = (float*)alloc((size_t)NN * 4 * 4);
  float*    f_dinv = (float*)alloc((size_t)NN * 4);
  float*    f_hp   = (float*)alloc((size_t)(N1 + 1) * 64 * 4);
  float*    f_h3   = (float*)alloc((size_t)N1 * 64 * 4);
  float*    f_h2   = (float*)alloc((size_t)GG * KP2 * 64 * 4);
  int*      i_perm = (int*)alloc((size_t)N1 * 4);
  int*      i_nid  = (int*)alloc((size_t)NN * 4);
  int*      i_s1   = (int*)alloc((size_t)EE * 4);
  int*      i_d1   = (int*)alloc((size_t)EE * 4);
  (void)ws_size; (void)in_sizes; (void)n_in; (void)out_size;

  auto ggrid = [](int M) { return (unsigned)((M + 127) / 128); };  // 8 waves x 16 rows

  // ---- GCN ----
  fill_f32_k<<<nb256(NN), 256, 0, stream>>>(f_dinv, 1.0f, NN);   // self-loop degree
  deg_kernel<<<nb256(EE), 256, 0, stream>>>(dst, f_dinv, EE);
  rsqrt_kernel<<<nb256(NN), 256, 0, stream>>>(f_dinv, NN);
  gemm_wmma<64, 64, false, false><<<ggrid(NN), 256, 0, stream>>>(x, nullptr, gcn_w, nullptr, f_hA, NN);
  fill_f32_k<<<nb256((size_t)NN * 64), 256, 0, stream>>>(f_h0, 0.0f, (size_t)NN * 64);
  gcn_scatter<<<nb256((size_t)EE * 64), 256, 0, stream>>>(src, dst, f_dinv, f_hA, f_h0, EE);
  gcn_self<<<nb256((size_t)NN * 64), 256, 0, stream>>>(f_dinv, f_hA, gcn_b, f_h0, NN);

  // ---- GATv2 #1 (H=4) ----
  gemm_wmma<64, 256, false, false><<<ggrid(NN), 256, 0, stream>>>(f_h0, nullptr, g1_wl, nullptr, big1, NN);
  gemm_wmma<64, 256, false, false><<<ggrid(NN), 256, 0, stream>>>(f_h0, nullptr, g1_wr, nullptr, big2, NN);
  fill_u32_k<<<nb256((size_t)NN * 4), 256, 0, stream>>>(u_smax, 0u, (size_t)NN * 4);
  fill_f32_k<<<nb256((size_t)NN * 4), 256, 0, stream>>>(f_ssum, 0.0f, (size_t)NN * 4);
  {
    size_t tot = (size_t)(EE + NN) * 4;
    gat_score<<<nb256(tot), 256, 0, stream>>>(src, dst, big1, big2, g1_att, f_sc, EE, NN, 4);
    seg_max_k<<<nb256(tot), 256, 0, stream>>>(dst, f_sc, u_smax, EE, NN, 4);
    seg_expsum_k<<<nb256(tot), 256, 0, stream>>>(dst, f_sc, u_smax, f_ssum, EE, NN, 4);
    fill_f32_k<<<nb256((size_t)NN * 256), 256, 0, stream>>>(big3, 0.0f, (size_t)NN * 256);
    seg_aggregate_k<<<nb256(tot * 64), 256, 0, stream>>>(src, dst, f_sc, f_ssum, big1, big3, EE, NN, 4);
  }
  // lin1: tanh(gat_out + g1_b) @ lin1_w + lin1_b
  gemm_wmma<256, 64, true, true><<<ggrid(NN), 256, 0, stream>>>(big3, g1_b, lin1_w, lin1_b, f_hA, NN);

  // ---- TransformerConv (H=2) ----
  float* q    = big1;
  float* kk   = big1 + (size_t)NN * 128;
  float* v    = big2;
  float* skip = big2 + (size_t)NN * 128;
  float* msg  = big3;
  float* tout = big3 + (size_t)NN * 128;
  gemm_wmma<64, 128, false, false><<<ggrid(NN), 256, 0, stream>>>(f_hA, nullptr, t_wq, nullptr, q, NN);
  gemm_wmma<64, 128, false, false><<<ggrid(NN), 256, 0, stream>>>(f_hA, nullptr, t_wk, nullptr, kk, NN);
  gemm_wmma<64, 128, false, false><<<ggrid(NN), 256, 0, stream>>>(f_hA, nullptr, t_wv, nullptr, v, NN);
  gemm_wmma<64, 128, false, false><<<ggrid(NN), 256, 0, stream>>>(f_hA, nullptr, t_wsk, t_bsk, skip, NN);
  fill_u32_k<<<nb256((size_t)NN * 4), 256, 0, stream>>>(u_smax, 0u, (size_t)NN * 4);
  fill_f32_k<<<nb256((size_t)NN * 4), 256, 0, stream>>>(f_ssum, 0.0f, (size_t)NN * 4);
  {
    size_t tot = (size_t)EE * 2;
    trans_score_k<<<nb256(tot), 256, 0, stream>>>(src, dst, q, kk, f_sc, EE, 2);
    seg_max_k<<<nb256(tot), 256, 0, stream>>>(dst, f_sc, u_smax, EE, 0, 2);
    seg_expsum_k<<<nb256(tot), 256, 0, stream>>>(dst, f_sc, u_smax, f_ssum, EE, 0, 2);
    fill_f32_k<<<nb256((size_t)NN * 128), 256, 0, stream>>>(msg, 0.0f, (size_t)NN * 128);
    seg_aggregate_k<<<nb256(tot * 64), 256, 0, stream>>>(src, dst, f_sc, f_ssum, v, msg, EE, 0, 2);
  }
  trans_beta_k<<<(NN + 7) / 8, 256, 0, stream>>>(msg, skip, t_wb, tout, NN);
  gemm_wmma<128, 64, true, false><<<ggrid(NN), 256, 0, stream>>>(tout, nullptr, lin2_w, lin2_b, f_hB, NN);

  // ---- GraphNorm ----
  graph_norm_k<<<GG, 64, 0, stream>>>(f_hB, gn_w, gn_b, gn_ms, f_hA, NPER);

  // ---- TopK pool 1 + edge filter ----
  topk_pool<512><<<GG, 512, 0, stream>>>(f_hA, p1_w, f_hp, i_perm, NPER, KP1);
  fill_f32_k<<<1, 256, 0, stream>>>(f_hp + (size_t)N1 * 64, 0.0f, 64);  // dummy sink row
  fill_i32_k<<<nb256(NN), 256, 0, stream>>>(i_nid, N1, NN);
  build_newid_k<<<nb256(N1), 256, 0, stream>>>(i_perm, i_nid, N1);
  remap_edges_k<<<nb256(EE), 256, 0, stream>>>(src, dst, i_nid, i_s1, i_d1, EE, N1);

  // ---- GATv2 #2 (H=2, n = N1+1 incl. dummy) ----
  const int n2 = N1 + 1;
  gemm_wmma<64, 128, false, false><<<ggrid(n2), 256, 0, stream>>>(f_hp, nullptr, g3_wl, nullptr, big1, n2);
  gemm_wmma<64, 128, false, false><<<ggrid(n2), 256, 0, stream>>>(f_hp, nullptr, g3_wr, nullptr, big2, n2);
  fill_u32_k<<<nb256((size_t)n2 * 2), 256, 0, stream>>>(u_smax, 0u, (size_t)n2 * 2);
  fill_f32_k<<<nb256((size_t)n2 * 2), 256, 0, stream>>>(f_ssum, 0.0f, (size_t)n2 * 2);
  {
    size_t tot = (size_t)(EE + n2) * 2;
    gat_score<<<nb256(tot), 256, 0, stream>>>(i_s1, i_d1, big1, big2, g3_att, f_sc, EE, n2, 2);
    seg_max_k<<<nb256(tot), 256, 0, stream>>>(i_d1, f_sc, u_smax, EE, n2, 2);
    seg_expsum_k<<<nb256(tot), 256, 0, stream>>>(i_d1, f_sc, u_smax, f_ssum, EE, n2, 2);
    fill_f32_k<<<nb256((size_t)n2 * 128), 256, 0, stream>>>(big3, 0.0f, (size_t)n2 * 128);
    seg_aggregate_k<<<nb256(tot * 64), 256, 0, stream>>>(i_s1, i_d1, f_sc, f_ssum, big1, big3, EE, n2, 2);
  }
  // lin3 over the first N1 rows: tanh(h3 + g3_b) @ lin3_w + lin3_b
  gemm_wmma<128, 64, true, true><<<ggrid(N1), 256, 0, stream>>>(big3, g3_b, lin3_w, lin3_b, f_h3, N1);

  // ---- TopK pool 2 + readout ----
  topk_pool<256><<<GG, 256, 0, stream>>>(f_h3, p2_w, f_h2, nullptr, KP1, KP2);
  readout_k<<<GG, 64, 0, stream>>>(f_h2, feats, KP2);
  final_out_k<<<1, 128, 0, stream>>>(feats, out_w, out_b, out, GG);
}
